// MultiHead_40415642255471
// MI455X (gfx1250) — matrix operand order repaired and checked by reading, not compiled, not run on hardware
//
#include <hip/hip_runtime.h>
#include <hip/hip_bf16.h>

// ---------------- problem constants ----------------
#define B_   2
#define T_   2048
#define C_   1024
#define H_   16
#define D_   64
#define SCALE 0.03125f   // C^-0.5 = 1/sqrt(1024)

typedef __bf16 bf16_t;
typedef __attribute__((ext_vector_type(16))) __bf16    v16bf;
typedef __attribute__((ext_vector_type(8)))  float     v8f;
typedef __attribute__((ext_vector_type(8)))  unsigned  v8u;

// ---------------- device helpers ----------------
__device__ __forceinline__ v8f wmma_bf16(v16bf a, v16bf b, v8f c) {
  // D = A(16x32 bf16) * B(32x16 bf16) + C(16x16 f32)
  return __builtin_amdgcn_wmma_f32_16x16x32_bf16(false, a, false, b, (short)0, c,
                                                 false, false);
}

__device__ __forceinline__ v8f zero8() {
  v8f z = {0.f, 0.f, 0.f, 0.f, 0.f, 0.f, 0.f, 0.f};
  return z;
}

// pack two f32 -> packed bf16 pair (lo = a, hi = b)
__device__ __forceinline__ unsigned pk2(float a, float b) {
  unsigned short lo = __builtin_bit_cast(unsigned short, (__bf16)a);
  unsigned short hi = __builtin_bit_cast(unsigned short, (__bf16)b);
  return (unsigned)lo | ((unsigned)hi << 16);
}

// load 16 bf16 from two 16B-aligned chunks into a fragment register
__device__ __forceinline__ v16bf load2x16(const bf16_t* pa, const bf16_t* pb) {
  uint4 a = *(const uint4*)pa;
  uint4 b = *(const uint4*)pb;
  v8u u;
  u[0] = a.x; u[1] = a.y; u[2] = a.z; u[3] = a.w;
  u[4] = b.x; u[5] = b.y; u[6] = b.z; u[7] = b.w;
  return __builtin_bit_cast(v16bf, u);
}

// A-fragment (16x32, 16-bit): lanes0-15 hold K[0..15]; lanes16-31 K[16..31]
__device__ __forceinline__ v16bf load_afrag(const bf16_t* row, int hi) {
  return load2x16(row + hi * 16, row + hi * 16 + 8);
}

// B-fragment (32x16, 16-bit): lane n holds column n; lanes0-15 K[0..15], lanes16-31 K[16..31]
__device__ __forceinline__ v16bf load_bfrag(const bf16_t* col, int hi) {
  return load2x16(col + hi * 16, col + hi * 16 + 8);
}

// cross-lane xor-16 (SWAPX16) via ds_swizzle, no LDS memory used
__device__ __forceinline__ float swz16f(float x) {
  int i = __builtin_bit_cast(int, x);
  i = __builtin_amdgcn_ds_swizzle(i, 0x401F);  // group-of-32: xor=0x10, and=0x1f
  return __builtin_bit_cast(float, i);
}
__device__ __forceinline__ unsigned swz16u(unsigned x) {
  int i = __builtin_amdgcn_ds_swizzle(__builtin_bit_cast(int, x), 0x401F);
  return __builtin_bit_cast(unsigned, i);
}

// ---------------- prologue: f32 -> bf16 conversions ----------------
__global__ void cvt_x_kernel(const float* __restrict__ x, bf16_t* __restrict__ xb) {
  int i = (blockIdx.x * blockDim.x + threadIdx.x) * 8;
  float4 a = ((const float4*)(x + i))[0];
  float4 b = ((const float4*)(x + i))[1];
  uint4 o;
  o.x = pk2(a.x, a.y); o.y = pk2(a.z, a.w);
  o.z = pk2(b.x, b.y); o.w = pk2(b.z, b.w);
  *(uint4*)(xb + i) = o;
}

// W [H,C,D] f32 -> Wt [H,D,C] bf16 (tiny: 3 x 1M elements, prologue only)
__global__ void cvt_w_kernel(const float* __restrict__ W, bf16_t* __restrict__ Wt) {
  int i = blockIdx.x * blockDim.x + threadIdx.x;       // < H*C*D
  int d = i % D_;
  int c = (i / D_) % C_;
  int h = i / (D_ * C_);
  Wt[(h * D_ + d) * C_ + c] = (bf16_t)W[i];
}

// ---------------- projection: Q^T,K^T (store [B,H,T,D]) and V (store Vt [B,H,D,T]) ----------------
__global__ void __launch_bounds__(128) proj_kernel(
    const bf16_t* __restrict__ xb,
    const bf16_t* __restrict__ Wqt, const bf16_t* __restrict__ Wkt,
    const bf16_t* __restrict__ Wvt,
    bf16_t* __restrict__ Q, bf16_t* __restrict__ K, bf16_t* __restrict__ Vt) {
  int wg = blockIdx.x;              // B*H*(T/64) = 1024
  int tb = wg & 31;                 // T/64 = 32 blocks
  int h  = (wg >> 5) & 15;
  int b  = wg >> 9;
  int wave = threadIdx.x >> 5;
  int lane = threadIdx.x & 31;
  int l15  = lane & 15;
  int hi   = lane >> 4;
  int t0   = tb * 64 + wave * 16;

  const bf16_t* xrow = xb + (b * T_ + t0 + l15) * C_;        // token row (lane = token)
  const bf16_t* wqr  = Wqt + (h * D_ + l15) * C_;            // feature rows (lane = d)
  const bf16_t* wkr  = Wkt + (h * D_ + l15) * C_;
  const bf16_t* wvr  = Wvt + (h * D_ + l15) * C_;

  v8f aq[4], ak[4], av[4];
#pragma unroll
  for (int i = 0; i < 4; ++i) { aq[i] = zero8(); ak[i] = zero8(); av[i] = zero8(); }

  for (int c0 = 0; c0 < C_; c0 += 32) {
    v16bf xB = load_bfrag(xrow + c0, hi);   // x^T as B (N=token, K=c)
    v16bf xA = load_afrag(xrow + c0, hi);   // x as A (M=token, K=c)
#pragma unroll
    for (int blk = 0; blk < 4; ++blk) {
      v16bf wq = load_afrag(wqr + blk * 16 * C_ + c0, hi);   // Wq^T as A (M=d)
      aq[blk] = wmma_bf16(wq, xB, aq[blk]);                  // Q^T tile
      v16bf wk = load_afrag(wkr + blk * 16 * C_ + c0, hi);
      ak[blk] = wmma_bf16(wk, xB, ak[blk]);                  // K^T tile
      v16bf wv = load_bfrag(wvr + blk * 16 * C_ + c0, hi);   // Wv as B (N=d, K=c)
      av[blk] = wmma_bf16(xA, wv, av[blk]);                  // V tile
    }
  }

  // Q^T/K^T accum: N(lane)=token, M(vgpr)=d  -> contiguous d store into [B,H,T,D]
  int qkbase = ((b * H_ + h) * T_ + t0 + l15) * D_ + hi * 8;
#pragma unroll
  for (int blk = 0; blk < 4; ++blk) {
    uint4 o;
    o.x = pk2(aq[blk][0], aq[blk][1]); o.y = pk2(aq[blk][2], aq[blk][3]);
    o.z = pk2(aq[blk][4], aq[blk][5]); o.w = pk2(aq[blk][6], aq[blk][7]);
    *(uint4*)(Q + qkbase + blk * 16) = o;
    o.x = pk2(ak[blk][0], ak[blk][1]); o.y = pk2(ak[blk][2], ak[blk][3]);
    o.z = pk2(ak[blk][4], ak[blk][5]); o.w = pk2(ak[blk][6], ak[blk][7]);
    *(uint4*)(K + qkbase + blk * 16) = o;
  }
  // V accum: N(lane)=d, M(vgpr)=token -> contiguous t store into Vt [B,H,D,T]
#pragma unroll
  for (int blk = 0; blk < 4; ++blk) {
    uint4 o;
    o.x = pk2(av[blk][0], av[blk][1]); o.y = pk2(av[blk][2], av[blk][3]);
    o.z = pk2(av[blk][4], av[blk][5]); o.w = pk2(av[blk][6], av[blk][7]);
    *(uint4*)(Vt + ((b * H_ + h) * D_ + blk * 16 + l15) * T_ + t0 + hi * 8) = o;
  }
}

// ---------------- flash attention: S^T = K Q^T, O^T = V^T P^T ----------------
__global__ void __launch_bounds__(128) attn_kernel(
    const bf16_t* __restrict__ Q, const bf16_t* __restrict__ K,
    const bf16_t* __restrict__ Vt, float* __restrict__ out) {
  int wg = blockIdx.x;              // B*H*(T/64)
  int tb = wg & 31;
  int h  = (wg >> 5) & 15;
  int b  = wg >> 9;
  int wave = threadIdx.x >> 5;
  int lane = threadIdx.x & 31;
  int l15  = lane & 15;
  int hi   = lane >> 4;
  int q0   = tb * 64 + wave * 16;
  int qi   = q0 + l15;              // this lane's query index

  const bf16_t* Kb  = K  + (b * H_ + h) * (T_ * D_);
  const bf16_t* Vtb = Vt + (b * H_ + h) * (D_ * T_);
  const bf16_t* qrow = Q + ((b * H_ + h) * T_ + qi) * D_;

  // Q^T as B-fragments (N = query, K = d), d split 0..31 / 32..63
  v16bf qb0 = load_bfrag(qrow + 0, hi);
  v16bf qb1 = load_bfrag(qrow + 32, hi);

  float m = -__builtin_inff();
  float l = 0.f;
  v8f o[4];
#pragma unroll
  for (int i = 0; i < 4; ++i) o[i] = zero8();

  for (int kb = 0; kb <= q0 + 15; kb += 32) {
    // S^T tile: M = key, N = query; two 16-key sub-blocks
    v8f s0 = zero8(), s1 = zero8();
    const bf16_t* kr0 = Kb + (kb + l15) * D_;
    const bf16_t* kr1 = Kb + (kb + 16 + l15) * D_;
    s0 = wmma_bf16(load_afrag(kr0 + 0, hi), qb0, s0);
    s0 = wmma_bf16(load_afrag(kr0 + 32, hi), qb1, s0);
    s1 = wmma_bf16(load_afrag(kr1 + 0, hi), qb0, s1);
    s1 = wmma_bf16(load_afrag(kr1 + 32, hi), qb1, s1);

    // scale + causal mask + per-lane key max (keys live in VGPRs!)
    float sv0[8], sv1[8];
    float cand = -__builtin_inff();
#pragma unroll
    for (int r = 0; r < 8; ++r) {
      int k0 = kb + r + hi * 8;
      float v0 = s0[r] * SCALE;
      v0 = (k0 > qi) ? -__builtin_inff() : v0;
      sv0[r] = v0; cand = fmaxf(cand, v0);
      int k1 = kb + 16 + r + hi * 8;
      float v1 = s1[r] * SCALE;
      v1 = (k1 > qi) ? -__builtin_inff() : v1;
      sv1[r] = v1; cand = fmaxf(cand, v1);
    }
    cand = fmaxf(cand, swz16f(cand));       // sync max across lane halves
    float mnew  = fmaxf(m, cand);
    float alpha = __expf(m - mnew);
    m = mnew;

    float p0[8], p1[8], psum = 0.f;
#pragma unroll
    for (int r = 0; r < 8; ++r) {
      p0[r] = __expf(sv0[r] - mnew);
      p1[r] = __expf(sv1[r] - mnew);
      psum += p0[r] + p1[r];
    }
    l = l * alpha + psum;
#pragma unroll
    for (int i = 0; i < 4; ++i) o[i] = o[i] * alpha;

    // Repack P^T (C-layout) into B-fragment layout: needs an xor-16 lane swap
    // C-layout : lanes0-15 keys[0..7]/[16..23] per frag, lanes16-31 keys[8..15]/[24..31]
    // B-layout : lanes0-15 keys[0..15],        lanes16-31 keys[16..31]
    unsigned A0[4], A1[4];
#pragma unroll
    for (int v = 0; v < 4; ++v) {
      A0[v] = pk2(p0[2 * v], p0[2 * v + 1]);
      A1[v] = pk2(p1[2 * v], p1[2 * v + 1]);
    }
    v8u pu;
#pragma unroll
    for (int v = 0; v < 4; ++v) {
      unsigned sw0 = swz16u(A0[v]);
      unsigned sw1 = swz16u(A1[v]);
      pu[v]     = hi ? sw1   : A0[v];   // keys 0..15 | 16..31
      pu[v + 4] = hi ? A1[v] : sw0;     // keys 8..15<->swap | 24..31
    }
    v16bf pb = __builtin_bit_cast(v16bf, pu);

    // O^T += V^T * P^T  (A = V^T tile, M=d rows; contiguous keys from Vt[D,T])
#pragma unroll
    for (int blk = 0; blk < 4; ++blk) {
      v16bf va = load_afrag(Vtb + (blk * 16 + l15) * T_ + kb, hi);
      o[blk] = wmma_bf16(va, pb, o[blk]);
    }
  }

  float ltot = l + swz16f(l);
  float inv  = 1.0f / ltot;

  // O^T accum: N(lane)=query, M(vgpr)=d -> contiguous f32 store into out [B,T,H*D]
  float* op = out + (b * T_ + qi) * (H_ * D_) + h * D_ + hi * 8;
#pragma unroll
  for (int blk = 0; blk < 4; ++blk) {
    float4 lo = {o[blk][0] * inv, o[blk][1] * inv, o[blk][2] * inv, o[blk][3] * inv};
    float4 hi4 = {o[blk][4] * inv, o[blk][5] * inv, o[blk][6] * inv, o[blk][7] * inv};
    *(float4*)(op + blk * 16) = lo;
    *(float4*)(op + blk * 16 + 4) = hi4;
  }
}

// ---------------- host launcher ----------------
extern "C" void kernel_launch(void* const* d_in, const int* in_sizes, int n_in,
                              void* d_out, int out_size, void* d_ws, size_t ws_size,
                              hipStream_t stream) {
  (void)in_sizes; (void)n_in; (void)out_size; (void)ws_size;
  const float* x  = (const float*)d_in[0];
  const float* Wq = (const float*)d_in[1];
  const float* Wk = (const float*)d_in[2];
  const float* Wv = (const float*)d_in[3];
  float* out = (float*)d_out;

  char* ws = (char*)d_ws;
  bf16_t* xb  = (bf16_t*)(ws);                       //  8,388,608 B : x in bf16 [B,T,C]
  bf16_t* Wqt = (bf16_t*)(ws + 8388608);             //  2,097,152 B : Wq^T bf16 [H,D,C]
  bf16_t* Wkt = Wqt + (H_ * D_ * C_);
  bf16_t* Wvt = Wkt + (H_ * D_ * C_);
  bf16_t* Qb  = (bf16_t*)(ws + 14680064);            //  8,388,608 B : Q bf16 [B,H,T,D]
  bf16_t* Kb  = Qb + (B_ * H_ * T_ * D_);            //  8,388,608 B : K bf16 [B,H,T,D]
  bf16_t* Vtb = Kb + (B_ * H_ * T_ * D_);            //  8,388,608 B : V^T bf16 [B,H,D,T]

  // prologue conversions
  cvt_x_kernel<<<(B_ * T_ * C_) / (256 * 8), 256, 0, stream>>>(x, xb);
  cvt_w_kernel<<<(H_ * C_ * D_) / 256, 256, 0, stream>>>(Wq, Wqt);
  cvt_w_kernel<<<(H_ * C_ * D_) / 256, 256, 0, stream>>>(Wk, Wkt);
  cvt_w_kernel<<<(H_ * C_ * D_) / 256, 256, 0, stream>>>(Wv, Wvt);

  // QKV projection (WMMA)
  int ngrp = B_ * H_ * (T_ / 64);  // 1024 workgroups, 4 waves each
  proj_kernel<<<ngrp, 128, 0, stream>>>(xb, Wqt, Wkt, Wvt, Qb, Kb, Vtb);

  // flash attention (WMMA)
  attn_kernel<<<ngrp, 128, 0, stream>>>(Qb, Kb, Vtb, out);
}
